// Attention1D_80607946211514
// MI455X (gfx1250) — matrix operand order repaired and checked by reading, not compiled, not run on hardware
//
#include <hip/hip_runtime.h>

// ---------------------------------------------------------------------------
// MI455X (gfx1250) fused GroupNorm + 1x1-conv QKV + flash attention + proj.
// v_wmma_f32_16x16x32_bf16 (wave32), fully-unrolled ping-pong GEMM mainloops,
// TDM (tensor_load_to_lds) double-buffered K/V staging in attention.
// ---------------------------------------------------------------------------

typedef __bf16 bf16;
typedef __bf16 v16bf __attribute__((ext_vector_type(16)));
typedef __bf16 v8bf  __attribute__((ext_vector_type(8)));
typedef float  v8f   __attribute__((ext_vector_type(8)));
typedef unsigned int u32x4 __attribute__((ext_vector_type(4)));
typedef int          i32x8 __attribute__((ext_vector_type(8)));
typedef int          i32x4 __attribute__((ext_vector_type(4)));

#define DEV __device__ __forceinline__

#if __has_builtin(__builtin_amdgcn_tensor_load_to_lds) && \
    __has_builtin(__builtin_amdgcn_s_wait_tensorcnt)
#define USE_TDM 1
#else
#define USE_TDM 0
#endif

constexpr int Bn = 4;       // batch
constexpr int Cn = 512;     // channels
constexpr int Hn = 8;       // heads
constexpr int Dn = 64;      // head dim
constexpr int Ln = 2048;    // sequence
constexpr int Gn = 32;      // groups
constexpr int CPG = Cn / Gn;
constexpr float EPS = 1e-5f;

DEV bf16 bf16_rne(float f) {
  unsigned u = __builtin_bit_cast(unsigned, f);
  u += 0x7fffu + ((u >> 16) & 1u);             // round-to-nearest-even
  unsigned short h = (unsigned short)(u >> 16);
  return __builtin_bit_cast(bf16, h);
}

DEV v8f wmma_bf16(v16bf a, v16bf b, v8f c) {
  return __builtin_amdgcn_wmma_f32_16x16x32_bf16(false, a, false, b,
                                                 (short)0, c, false, false);
}

// A fragment 16(M)x32(K): lane m<16 holds row m, K = kk+{0..7, 16..23};
// lane m+16 holds row m, K = kk+{8..15, 24..31}. Two 16B contiguous loads.
DEV v16bf load_frag_a(const bf16* __restrict__ base, int ld, int m0, int kk,
                      int lane) {
  int m = lane & 15, hs = lane >> 4;
  const bf16* p = base + (size_t)(m0 + m) * ld + (kk + 8 * hs);
  v8bf lo = *(const v8bf*)(p);
  v8bf hi = *(const v8bf*)(p + 16);
  v16bf r;
#pragma unroll
  for (int i = 0; i < 8; ++i) { r[i] = lo[i]; r[i + 8] = hi[i]; }
  return r;
}

// B fragment 32(K)x16(N), operand stored K-contiguous as Bt[N][K]:
// lane n<16 holds col n, K = kk+{0..7, 16..23}; lane n+16 holds K = kk+{8..15, 24..31}.
DEV v16bf load_frag_b(const bf16* __restrict__ base, int ld, int n0, int kk,
                      int lane) {
  int n = lane & 15, hs = lane >> 4;
  const bf16* p = base + (size_t)(n0 + n) * ld + (kk + 8 * hs);
  v8bf lo = *(const v8bf*)(p);
  v8bf hi = *(const v8bf*)(p + 16);
  v16bf r;
#pragma unroll
  for (int i = 0; i < 8; ++i) { r[i] = lo[i]; r[i + 8] = hi[i]; }
  return r;
}

#if USE_TDM
// Issue a 2D TDM tile load (bf16 elements) into LDS. D# group layout per
// CDNA5 ISA 8.3/8.4; groups 2/3 zero (2D tensor). Wave-uniform operands.
// 6-arg builtin signature (clang-23 / therock-10.0 headers).
DEV void tdm_load_2d(unsigned lds_off, unsigned long long gaddr,
                     unsigned dim0, unsigned dim1, unsigned stride0,
                     unsigned tile0, unsigned tile1) {
  u32x4 g0;
  g0[0] = 1u;                                          // count=1, no gather
  g0[1] = lds_off;                                     // LDS byte address
  g0[2] = (unsigned)gaddr;                             // global_addr[31:0]
  g0[3] = (unsigned)((gaddr >> 32) & 0x01FFFFFFull) |  // global_addr[56:32]
          (2u << 30);                                  // type = 2 (image)
  i32x8 g1;
  g1[0] = (int)(1u << 16);                             // data_size = 2 bytes
  g1[1] = (int)((dim0 & 0xFFFFu) << 16);               // tensor_dim0[15:0]
  g1[2] = (int)((dim0 >> 16) | ((dim1 & 0xFFFFu) << 16));
  g1[3] = (int)((dim1 >> 16) | (tile0 << 16));         // tile_dim0
  g1[4] = (int)tile1;                                  // tile_dim1 (dim2=0)
  g1[5] = (int)stride0;                                // dim0_stride[31:0]
  g1[6] = 0;
  g1[7] = 0;
  i32x4 z4 = {0, 0, 0, 0};
  i32x8 z8 = {0, 0, 0, 0, 0, 0, 0, 0};
  __builtin_amdgcn_tensor_load_to_lds(g0, g1, z4, z4, z8, 0);
}
#endif

// ---------------------------------------------------------------------------
// Kernel 1: fp32 -> bf16 weight conversion
// ---------------------------------------------------------------------------
__global__ void k_f32_to_bf16(const float* __restrict__ in,
                              bf16* __restrict__ out, int n) {
  int i = blockIdx.x * 256 + threadIdx.x;
  if (i < n) out[i] = bf16_rne(in[i]);
}

// ---------------------------------------------------------------------------
// Kernel 2: GroupNorm over (CPG, L) per (b,g); writes h^T [B, L, C] bf16.
// ---------------------------------------------------------------------------
__global__ __launch_bounds__(256) void k_groupnorm(
    const float* __restrict__ x, const float* __restrict__ gamma,
    const float* __restrict__ beta, bf16* __restrict__ ht) {
  int b = blockIdx.x >> 5, g = blockIdx.x & 31;
  const float4* xp = (const float4*)(x + ((size_t)b * Cn + g * CPG) * Ln);
  const int NV4 = CPG * (Ln / 4);                 // 8192 float4
  float s = 0.f, s2 = 0.f;
  for (int i = threadIdx.x; i < NV4; i += 256) {
    float4 v = xp[i];
    s  += v.x + v.y + v.z + v.w;
    s2 += v.x * v.x + v.y * v.y + v.z * v.z + v.w * v.w;
  }
#pragma unroll
  for (int off = 16; off; off >>= 1) {
    s  += __shfl_xor(s, off, 32);
    s2 += __shfl_xor(s2, off, 32);
  }
  __shared__ float ls[8], ls2[8];
  int lane = threadIdx.x & 31, wave = threadIdx.x >> 5;
  if (lane == 0) { ls[wave] = s; ls2[wave] = s2; }
  __syncthreads();
  float ts = 0.f, ts2 = 0.f;
#pragma unroll
  for (int w = 0; w < 8; ++w) { ts += ls[w]; ts2 += ls2[w]; }
  const float inv_n = 1.0f / (float)(CPG * Ln);
  float mean = ts * inv_n;
  float var  = ts2 * inv_n - mean * mean;
  float rstd = rsqrtf(var + EPS);

  for (int i = threadIdx.x; i < NV4; i += 256) {
    int cl = i >> 9, l4 = i & 511;
    int c = g * CPG + cl, l = l4 * 4;
    float4 v = xp[i];
    float ga = gamma[c] * rstd, be = beta[c] - mean * gamma[c] * rstd;
    bf16* w = ht + ((size_t)b * Ln + l) * Cn + c;   // transposed scatter
    w[0 * Cn] = bf16_rne(v.x * ga + be);
    w[1 * Cn] = bf16_rne(v.y * ga + be);
    w[2 * Cn] = bf16_rne(v.z * ga + be);
    w[3 * Cn] = bf16_rne(v.w * ga + be);
  }
}

// ---------------------------------------------------------------------------
// Fully-unrolled ping-pong GEMM mainloop: wave tile 16x64, K-step 32, two
// alternating fragment register sets (no rotation moves), loads for stage
// k+1 issued before the WMMAs of stage k so waits stay partial.
// ---------------------------------------------------------------------------
DEV void gemm_mainloop(const bf16* __restrict__ A, const bf16* __restrict__ Bt,
                       int m0, int n0, int lane, v8f acc[4]) {
  v16bf a0 = load_frag_a(A, Cn, m0, 0, lane);
  v16bf b0[4];
#pragma unroll
  for (int t = 0; t < 4; ++t) b0[t] = load_frag_b(Bt, Cn, n0 + 16 * t, 0, lane);
#pragma unroll
  for (int kk = 0; kk < Cn; kk += 64) {
    v16bf a1 = load_frag_a(A, Cn, m0, kk + 32, lane);
    v16bf b1[4];
#pragma unroll
    for (int t = 0; t < 4; ++t)
      b1[t] = load_frag_b(Bt, Cn, n0 + 16 * t, kk + 32, lane);
#pragma unroll
    for (int t = 0; t < 4; ++t) acc[t] = wmma_bf16(a0, b0[t], acc[t]);
    if (kk + 64 < Cn) {
      a0 = load_frag_a(A, Cn, m0, kk + 64, lane);
#pragma unroll
      for (int t = 0; t < 4; ++t)
        b0[t] = load_frag_b(Bt, Cn, n0 + 16 * t, kk + 64, lane);
    }
#pragma unroll
    for (int t = 0; t < 4; ++t) acc[t] = wmma_bf16(a1, b1[t], acc[t]);
  }
}

// ---------------------------------------------------------------------------
// Kernel 3: QKV GEMM (3C x C)*(C x L) per batch; epilogue adds bias and
// scatters into Q,K [B,H,L,D] and V^T [B,H,D,L].
// ---------------------------------------------------------------------------
__global__ __launch_bounds__(256) void k_qkv_gemm(
    const bf16* __restrict__ wq, const bf16* __restrict__ ht,
    const float* __restrict__ bias, bf16* __restrict__ Q,
    bf16* __restrict__ K, bf16* __restrict__ Vt) {
  int b = blockIdx.z;
  int lane = threadIdx.x & 31, wave = threadIdx.x >> 5;
  int m0 = blockIdx.x * 128 + wave * 16;
  int n0 = blockIdx.y * 64;
  const bf16* Bm = ht + (size_t)b * Ln * Cn;
  v8f acc[4] = {};
  gemm_mainloop(wq, Bm, m0, n0, lane, acc);

  int hs = lane >> 4, nl = lane & 15;
#pragma unroll
  for (int r = 0; r < 8; ++r) {
    int o = m0 + r + 8 * hs;
    float bo = bias[o];
#pragma unroll
    for (int t = 0; t < 4; ++t) {
      int l = n0 + 16 * t + nl;
      bf16 v = bf16_rne(acc[t][r] + bo);
      if (o < Cn) {
        int oh = o;
        Q[(((size_t)b * Hn + (oh >> 6)) * Ln + l) * Dn + (oh & 63)] = v;
      } else if (o < 2 * Cn) {
        int oh = o - Cn;
        K[(((size_t)b * Hn + (oh >> 6)) * Ln + l) * Dn + (oh & 63)] = v;
      } else {
        int oh = o - 2 * Cn;
        Vt[(((size_t)b * Hn + (oh >> 6)) * Dn + (oh & 63)) * Ln + l] = v;
      }
    }
  }
}

// ---------------------------------------------------------------------------
// Kernel 4: flash attention. One wave = 16 queries; S^T = K*Q^T so score C/D
// layout == P A-fragment layout (P stays in registers). With TDM: one wave
// per block DMAs the shared 32-key K / V^T tiles into double-buffered LDS
// (tensor_load_to_lds + s_wait_tensorcnt), overlapping DMA with WMMA/softmax.
// ---------------------------------------------------------------------------
__global__ __launch_bounds__(256) void k_attention(
    const bf16* __restrict__ Q, const bf16* __restrict__ K,
    const bf16* __restrict__ Vt, bf16* __restrict__ ot) {
  int lane = threadIdx.x & 31, wave = threadIdx.x >> 5;
  int tile = blockIdx.x * 8 + wave;           // [0, B*H*L/16)
  int bh = tile >> 7;
  int q0 = (tile & 127) * 16;
  int b = bh >> 3, h = bh & 7;
  const bf16* Qb = Q  + (size_t)bh * Ln * Dn;
  const bf16* Kb = K  + (size_t)bh * Ln * Dn;
  const bf16* Vb = Vt + (size_t)bh * Dn * Ln;
  const float scale = 0.125f;

  v16bf qf0 = load_frag_b(Qb, Dn, q0, 0, lane);   // loop-invariant Q^T frags
  v16bf qf1 = load_frag_b(Qb, Dn, q0, 32, lane);

  v8f acc[4] = {};
  float mrun = -3.0e38f, lrun = 0.f;

#if USE_TDM
  __shared__ bf16 kbuf[2][32 * 64];               // [key][d], 4KB each
  __shared__ bf16 vbuf[2][64 * 32];               // [d][key], 4KB each
  unsigned long long kga = (unsigned long long)(size_t)Kb;
  unsigned long long vga = (unsigned long long)(size_t)Vb;
  if (wave == 0) {                                // prologue: tile 0 -> buf 0
    tdm_load_2d((unsigned)(size_t)&kbuf[0][0], kga, Dn, Ln, Dn, Dn, 32);
    tdm_load_2d((unsigned)(size_t)&vbuf[0][0], vga, Ln, Dn, Ln, 32, Dn);
  }
#endif

  for (int it = 0; it < Ln / 32; ++it) {
    int k0 = it * 32;
#if USE_TDM
    int cb = it & 1, nb = cb ^ 1;
    __syncthreads();                              // readers done with buf nb
    if (wave == 0) {
      if (it + 1 < Ln / 32) {
        int kn = k0 + 32;
        tdm_load_2d((unsigned)(size_t)&kbuf[nb][0],
                    kga + (unsigned long long)kn * Dn * 2, Dn, Ln, Dn, Dn, 32);
        tdm_load_2d((unsigned)(size_t)&vbuf[nb][0],
                    vga + (unsigned long long)kn * 2, Ln, Dn, Ln, 32, Dn);
        __builtin_amdgcn_s_wait_tensorcnt(2);     // tile `it` pair complete
      } else {
        __builtin_amdgcn_s_wait_tensorcnt(0);
      }
    }
    __syncthreads();                              // buf cb ready for all waves
    const bf16* kt = &kbuf[cb][0];
    const bf16* vt = &vbuf[cb][0];
    v16bf ka00 = load_frag_a(kt, Dn, 0, 0, lane);   // ds_load_b128 pairs
    v16bf ka01 = load_frag_a(kt, Dn, 0, 32, lane);
    v16bf ka10 = load_frag_a(kt, Dn, 16, 0, lane);
    v16bf ka11 = load_frag_a(kt, Dn, 16, 32, lane);
#else
    v16bf ka00 = load_frag_a(Kb, Dn, k0, 0, lane);
    v16bf ka01 = load_frag_a(Kb, Dn, k0, 32, lane);
    v16bf ka10 = load_frag_a(Kb, Dn, k0 + 16, 0, lane);
    v16bf ka11 = load_frag_a(Kb, Dn, k0 + 16, 32, lane);
#endif
    v8f st0 = {}; st0 = wmma_bf16(ka00, qf0, st0); st0 = wmma_bf16(ka01, qf1, st0);
    v8f st1 = {}; st1 = wmma_bf16(ka10, qf0, st1); st1 = wmma_bf16(ka11, qf1, st1);
    st0 *= scale; st1 *= scale;

    float pm = -3.0e38f;
#pragma unroll
    for (int r = 0; r < 8; ++r) pm = fmaxf(pm, fmaxf(st0[r], st1[r]));
    pm = fmaxf(pm, __shfl_xor(pm, 16, 32));
    float mnew  = fmaxf(mrun, pm);
    float alpha = __expf(mrun - mnew);
    mrun = mnew;

    v16bf pf; float ps = 0.f;
#pragma unroll
    for (int r = 0; r < 8; ++r) {
      float p0 = __expf(st0[r] - mnew);
      float p1 = __expf(st1[r] - mnew);
      ps += p0 + p1;
      pf[r] = bf16_rne(p0);
      pf[r + 8] = bf16_rne(p1);
    }
    ps += __shfl_xor(ps, 16, 32);
    lrun = lrun * alpha + ps;

    int hs8 = (lane >> 4) * 8;
#pragma unroll
    for (int r = 0; r < 8; ++r) {
      float ar = __shfl(alpha, r + hs8, 32);
#pragma unroll
      for (int t = 0; t < 4; ++t) acc[t][r] *= ar;
    }
#pragma unroll
    for (int t = 0; t < 4; ++t) {
#if USE_TDM
      v16bf vf = load_frag_b(vt, 32, 16 * t, 0, lane);
#else
      v16bf vf = load_frag_b(Vb, Ln, 16 * t, k0, lane);
#endif
      acc[t] = wmma_bf16(pf, vf, acc[t]);
    }
  }

  float linv = 1.0f / lrun;
  int hs8 = (lane >> 4) * 8, nl = lane & 15;
  bf16* op = ot + (size_t)b * Ln * Cn + (size_t)h * Dn;
#pragma unroll
  for (int r = 0; r < 8; ++r) {
    float lr = __shfl(linv, r + hs8, 32);
    int q = q0 + r + hs8;
#pragma unroll
    for (int t = 0; t < 4; ++t)
      op[(size_t)q * Cn + 16 * t + nl] = bf16_rne(acc[t][r] * lr);
  }
}

// ---------------------------------------------------------------------------
// Kernel 5: projection GEMM (C x C)*(C x L) + bias + residual, fp32 out.
// ---------------------------------------------------------------------------
__global__ __launch_bounds__(256) void k_proj_gemm(
    const bf16* __restrict__ wp, const bf16* __restrict__ ot,
    const float* __restrict__ bias, const float* __restrict__ x,
    float* __restrict__ out) {
  int b = blockIdx.z;
  int lane = threadIdx.x & 31, wave = threadIdx.x >> 5;
  int m0 = blockIdx.x * 128 + wave * 16;
  int n0 = blockIdx.y * 64;
  const bf16* Bm = ot + (size_t)b * Ln * Cn;
  v8f acc[4] = {};
  gemm_mainloop(wp, Bm, m0, n0, lane, acc);

  int hs = lane >> 4, nl = lane & 15;
#pragma unroll
  for (int r = 0; r < 8; ++r) {
    int o = m0 + r + 8 * hs;
    float bo = bias[o];
#pragma unroll
    for (int t = 0; t < 4; ++t) {
      int l = n0 + 16 * t + nl;
      size_t idx = ((size_t)b * Cn + o) * Ln + l;
      out[idx] = x[idx] + acc[t][r] + bo;
    }
  }
}

// ---------------------------------------------------------------------------
extern "C" void kernel_launch(void* const* d_in, const int* in_sizes, int n_in,
                              void* d_out, int out_size, void* d_ws,
                              size_t ws_size, hipStream_t stream) {
  (void)in_sizes; (void)n_in; (void)out_size; (void)ws_size;
  const float* x      = (const float*)d_in[0];
  const float* gamma  = (const float*)d_in[1];
  const float* beta   = (const float*)d_in[2];
  const float* w_qkv  = (const float*)d_in[3];
  const float* b_qkv  = (const float*)d_in[4];
  const float* w_proj = (const float*)d_in[5];
  const float* b_proj = (const float*)d_in[6];
  float* out = (float*)d_out;

  char* ws = (char*)d_ws;
  const size_t SZ_BLC = (size_t)Bn * Ln * Cn * sizeof(bf16);   // 8 MB
  bf16* ht  = (bf16*)(ws);                       // h^T; reused as o^T later
  bf16* Qb  = (bf16*)(ws + SZ_BLC);
  bf16* Kb  = (bf16*)(ws + 2 * SZ_BLC);
  bf16* Vtb = (bf16*)(ws + 3 * SZ_BLC);
  bf16* wqb = (bf16*)(ws + 4 * SZ_BLC);
  bf16* wpb = (bf16*)(ws + 4 * SZ_BLC + (size_t)3 * Cn * Cn * sizeof(bf16));

  k_f32_to_bf16<<<(3 * Cn * Cn + 255) / 256, 256, 0, stream>>>(w_qkv, wqb,
                                                               3 * Cn * Cn);
  k_f32_to_bf16<<<(Cn * Cn + 255) / 256, 256, 0, stream>>>(w_proj, wpb,
                                                           Cn * Cn);
  k_groupnorm<<<Bn * Gn, 256, 0, stream>>>(x, gamma, beta, ht);
  k_qkv_gemm<<<dim3(3 * Cn / 128, Ln / 64, Bn), 256, 0, stream>>>(
      wqb, ht, b_qkv, Qb, Kb, Vtb);
  k_attention<<<Bn * Hn * (Ln / 16) / 8, 256, 0, stream>>>(Qb, Kb, Vtb, ht);
  k_proj_gemm<<<dim3(Cn / 128, Ln / 64, Bn), 256, 0, stream>>>(
      wpb, ht, b_proj, x, out);
}
